// HybridSignToTextModel_38878043963663
// MI455X (gfx1250) — matrix operand order repaired and checked by reading, not compiled, not run on hardware
//
#include <hip/hip_runtime.h>
#include <hip/hip_bf16.h>

// ---------------------------------------------------------------------------
// Types for CDNA5 WMMA (wave32, v_wmma_f32_16x16x32_bf16)
// ---------------------------------------------------------------------------
typedef __bf16 bf16_t;
typedef __attribute__((ext_vector_type(8)))  __bf16 v8bf;
typedef __attribute__((ext_vector_type(16))) __bf16 v16bf;
typedef __attribute__((ext_vector_type(8)))  float  v8f;

#define GF_RELU 1
#define GF_MASK 2   // zero rows where (row % Sdim) >= nl[row / Sdim]

// Model constants
#define BSZ   64
#define LSRC  512
#define DIN   432
#define CC    256
#define S1    256
#define S2    128
#define HH    512
#define ENC   512
#define DH    256
#define EMB   256
#define ATT   128
#define GV    100
#define TV    200
#define TDEC  50
#define K1PAD 2176   // 5*432=2160 padded to 32
#define K2    1280   // 5*256

// ---------------------------------------------------------------------------
// Generic bf16 WMMA GEMM: C[M,N] = A[M,K] @ W[N,K]^T (+bias, +relu, +mask)
// A row-major bf16 (lda), W row-major bf16 (ldw = K stride).
// Each wave computes a 32x32 tile via 2x2 v_wmma_f32_16x16x32_bf16 fragments.
// M must be a multiple of 32; N, arbitrary (guarded); K multiple of 32.
// ---------------------------------------------------------------------------
__global__ __launch_bounds__(256)
void k_gemm(const bf16_t* __restrict__ A, int lda,
            const bf16_t* __restrict__ W, int ldw,
            const float* __restrict__ bias,
            float* __restrict__ Cf, int ldcf,
            bf16_t* __restrict__ Cb, int ldcb,
            int M, int N, int Kd, int flags,
            const int* __restrict__ nl, int Sdim)
{
    const int lane = threadIdx.x & 31;
    const int wave = threadIdx.x >> 5;
    const int tilesN = (N + 31) >> 5;
    const int tilesM = (M + 31) >> 5;
    int tile = blockIdx.x * 8 + wave;
    if (tile >= tilesM * tilesN) return;
    const int tm = tile / tilesN;
    const int tn = tile - tm * tilesN;
    const int row0 = tm * 32;
    const int col0 = tn * 32;

    const int m  = lane & 15;
    const int hi = lane >> 4;

    // A fragment (16x32 bf16): lane(m,hi) holds K[hi*8 .. hi*8+7] and
    // K[hi*8+16 .. hi*8+23] of row m  -> two 16B loads per fragment.
    const bf16_t* a0 = A + (size_t)(row0 + m)      * lda + hi * 8;
    const bf16_t* a1 = A + (size_t)(row0 + 16 + m) * lda + hi * 8;
    // B fragment (32x16 bf16): lane(n,hi) holds K[hi*8 .. hi*8+7] and
    // K[hi*8+16 .. hi*8+23] of column n (= weight row n) -> two 16B loads per fragment.
    int wr0 = col0 + m;      if (wr0 >= N) wr0 = N - 1;
    int wr1 = col0 + 16 + m; if (wr1 >= N) wr1 = N - 1;
    const bf16_t* b0 = W + (size_t)wr0 * ldw + hi * 8;
    const bf16_t* b1 = W + (size_t)wr1 * ldw + hi * 8;

    v8f acc00 = {}, acc01 = {}, acc10 = {}, acc11 = {};

    for (int k = 0; k < Kd; k += 32) {
        __builtin_prefetch(a0 + k + 128, 0, 1);
        __builtin_prefetch(b0 + k + 128, 0, 1);
        union { v16bf v; v8bf h[2]; } fA0, fA1, fB0, fB1;
        fA0.h[0] = *(const v8bf*)(a0 + k);
        fA0.h[1] = *(const v8bf*)(a0 + k + 16);
        fA1.h[0] = *(const v8bf*)(a1 + k);
        fA1.h[1] = *(const v8bf*)(a1 + k + 16);
        fB0.h[0] = *(const v8bf*)(b0 + k);
        fB0.h[1] = *(const v8bf*)(b0 + k + 16);
        fB1.h[0] = *(const v8bf*)(b1 + k);
        fB1.h[1] = *(const v8bf*)(b1 + k + 16);
        acc00 = __builtin_amdgcn_wmma_f32_16x16x32_bf16(false, fA0.v, false, fB0.v,
                                                        (short)0, acc00, false, false);
        acc01 = __builtin_amdgcn_wmma_f32_16x16x32_bf16(false, fA0.v, false, fB1.v,
                                                        (short)0, acc01, false, false);
        acc10 = __builtin_amdgcn_wmma_f32_16x16x32_bf16(false, fA1.v, false, fB0.v,
                                                        (short)0, acc10, false, false);
        acc11 = __builtin_amdgcn_wmma_f32_16x16x32_bf16(false, fA1.v, false, fB1.v,
                                                        (short)0, acc11, false, false);
    }

    // Epilogue. C fragment: element v -> row = v + 8*hi, col = lane&15.
    v8f accs[2][2] = { { acc00, acc01 }, { acc10, acc11 } };
    for (int i = 0; i < 2; ++i) {
        for (int j = 0; j < 2; ++j) {
            int col = col0 + 16 * j + m;
            if (col >= N) continue;
            float bv = bias ? bias[col] : 0.0f;
            for (int v = 0; v < 8; ++v) {
                int row = row0 + 16 * i + hi * 8 + v;
                float val = accs[i][j][v] + bv;
                if (flags & GF_RELU) val = val > 0.f ? val : 0.f;
                if (flags & GF_MASK) {
                    int bb = row / Sdim;
                    int tt = row - bb * Sdim;
                    if (tt >= nl[bb]) val = 0.f;
                }
                if (Cf) Cf[(size_t)row * ldcf + col] = val;
                if (Cb) Cb[(size_t)row * ldcb + col] = (bf16_t)val;
            }
        }
    }
}

// ---------------------------------------------------------------------------
// Prep kernels
// ---------------------------------------------------------------------------
__global__ void k_cvt(bf16_t* __restrict__ dst, const float* __restrict__ src, int n)
{
    int i = blockIdx.x * 256 + threadIdx.x;
    if (i < n) dst[i] = (bf16_t)src[i];
}

__global__ void k_zero_f32(float* __restrict__ p, int n)
{
    int i = blockIdx.x * 256 + threadIdx.x;
    if (i < n) p[i] = 0.f;
}

__global__ void k_zero_bf(bf16_t* __restrict__ p, int n)
{
    int i = blockIdx.x * 256 + threadIdx.x;
    if (i < n) p[i] = (bf16_t)0.f;
}

// Fold BatchNorm into conv weight/bias; repack weight [C, Cin, 5] -> [C, Kpad]
// with kk = k*Cin + i (matches im2col layout below).
__global__ void k_fold_conv(bf16_t* __restrict__ dW, float* __restrict__ dB,
                            const float* __restrict__ w, const float* __restrict__ cb,
                            const float* __restrict__ g, const float* __restrict__ be,
                            const float* __restrict__ mean, const float* __restrict__ var,
                            int Cin, int Kpad)
{
    int c = blockIdx.x;
    float sc = g[c] * rsqrtf(var[c] + 1e-5f);
    if (threadIdx.x == 0) dB[c] = (cb[c] - mean[c]) * sc + be[c];
    int Kr = Cin * 5;
    for (int kk = threadIdx.x; kk < Kpad; kk += blockDim.x) {
        float v = 0.f;
        if (kk < Kr) {
            int k = kk / Cin;
            int i = kk - k * Cin;
            v = w[((size_t)c * Cin + i) * 5 + k] * sc;
        }
        dW[(size_t)c * Kpad + kk] = (bf16_t)v;
    }
}

// nl = clip((nl-1)//2 + 1, 1, 128) twice
__global__ void k_nl(int* __restrict__ nl, const int* __restrict__ lengths)
{
    int b = threadIdx.x;
    if (b < BSZ) {
        int v = lengths[b];
        for (int r = 0; r < 2; ++r) {
            v = (v + 4 - 5) / 2 + 1;
            if (v < 1) v = 1;
            if (v > S2) v = S2;
        }
        nl[b] = v;
    }
}

// ---------------------------------------------------------------------------
// im2col (conv as GEMM).  x[b,i,l] = src[b,l,i].  A[(b,t)][k*Cin+i] = x[b,i,2t+k-2]
// ---------------------------------------------------------------------------
__global__ void k_im2col1(bf16_t* __restrict__ dst, const float* __restrict__ src)
{
    int row = blockIdx.x;          // b*S1 + t
    int b = row >> 8;
    int t = row & (S1 - 1);
    bf16_t* d = dst + (size_t)row * K1PAD;
    for (int kk = threadIdx.x; kk < K1PAD; kk += blockDim.x) {
        float v = 0.f;
        if (kk < DIN * 5) {
            int k = kk / DIN;
            int i = kk - k * DIN;
            int l = t * 2 + k - 2;
            if (l >= 0 && l < LSRC) v = src[((size_t)b * LSRC + l) * DIN + i];
        }
        d[kk] = (bf16_t)v;
    }
}

__global__ void k_im2col2(bf16_t* __restrict__ dst, const bf16_t* __restrict__ X1)
{
    int row = blockIdx.x;          // b*S2 + t
    int b = row >> 7;
    int t = row & (S2 - 1);
    bf16_t* d = dst + (size_t)row * K2;
    for (int kk = threadIdx.x; kk < K2; kk += blockDim.x) {
        int k = kk >> 8;           // /256
        int c = kk & 255;
        int l = t * 2 + k - 2;
        bf16_t v = (bf16_t)0.f;
        if (l >= 0 && l < S1) v = X1[((size_t)b * S1 + l) * CC + c];
        d[kk] = v;
    }
}

// ---------------------------------------------------------------------------
// Encoder GRU pointwise (both directions in one launch).
// ---------------------------------------------------------------------------
__global__ void k_gru_pw(float* __restrict__ Hf, float* __restrict__ Hb,
                         bf16_t* __restrict__ Hfb, bf16_t* __restrict__ Hbb,
                         bf16_t* __restrict__ hout,       // [B*S, 2H]
                         const float* __restrict__ GIf, const float* __restrict__ GIb,
                         const float* __restrict__ GHf, const float* __restrict__ GHb,
                         int t)
{
    int idx = blockIdx.x * blockDim.x + threadIdx.x;   // 2*64*512
    if (idx >= 2 * BSZ * HH) return;
    int dir = idx >> 15;
    int r   = idx & 32767;
    int b   = r >> 9;
    int j   = r & (HH - 1);
    int teff = dir ? (S2 - 1 - t) : t;
    const float* GI = dir ? GIb : GIf;
    const float* GH = dir ? GHb : GHf;
    float*  H   = dir ? Hb  : Hf;
    bf16_t* Hbf = dir ? Hbb : Hfb;

    const float* gi = GI + ((size_t)b * S2 + teff) * (3 * HH);
    const float* gh = GH + (size_t)b * (3 * HH);
    float ir = gi[j], iz = gi[HH + j], in_ = gi[2 * HH + j];
    float hr = gh[j], hz = gh[HH + j], hn  = gh[2 * HH + j];
    float rg = 1.f / (1.f + __expf(-(ir + hr)));
    float zg = 1.f / (1.f + __expf(-(iz + hz)));
    float ng = tanhf(in_ + rg * hn);
    float hp = H[(size_t)b * HH + j];
    float hnew = (1.f - zg) * ng + zg * hp;
    H[(size_t)b * HH + j] = hnew;
    Hbf[(size_t)b * HH + j] = (bf16_t)hnew;
    hout[((size_t)b * S2 + teff) * (2 * HH) + dir * HH + j] = (bf16_t)hnew;
}

// ---------------------------------------------------------------------------
// Decoder embedding gather: dec_in = [SOS, tgt[:, :-1]]
// ---------------------------------------------------------------------------
__global__ void k_emb(bf16_t* __restrict__ emb_all, const int* __restrict__ tgt,
                      const bf16_t* __restrict__ emb)
{
    int idx = blockIdx.x * 256 + threadIdx.x;      // B*TDEC*EMB
    if (idx >= BSZ * TDEC * EMB) return;
    int c = idx & (EMB - 1);
    int r = idx >> 8;
    int t = r % TDEC;
    int b = r / TDEC;
    int tok = (t == 0) ? 1 : tgt[b * TDEC + t - 1];
    emb_all[idx] = emb[(size_t)tok * EMB + c];
}

// ---------------------------------------------------------------------------
// Attention + context + xg assembly for one decoder step. grid=B, block=128.
// ---------------------------------------------------------------------------
__global__ void k_attn(const float* __restrict__ ENCP, const float* __restrict__ DP,
                       const float* __restrict__ enc, const float* __restrict__ av,
                       const int* __restrict__ nl, const bf16_t* __restrict__ emb_all,
                       bf16_t* __restrict__ xg, float* __restrict__ ctx, int t)
{
    __shared__ float sv[S2];
    __shared__ float ss[S2];
    int b   = blockIdx.x;
    int tid = threadIdx.x;     // == s index, 0..127

    const float* dp = DP + b * ATT;
    const float* ep = ENCP + ((size_t)b * S2 + tid) * ATT;
    float acc = 0.f;
    for (int a = 0; a < ATT; ++a) acc += tanhf(ep[a] + dp[a]) * av[a];
    float e = (tid < nl[b]) ? acc : -3.0e38f;

    ss[tid] = e;
    __syncthreads();
    for (int off = 64; off > 0; off >>= 1) {
        if (tid < off) ss[tid] = fmaxf(ss[tid], ss[tid + off]);
        __syncthreads();
    }
    float mx = ss[0];
    __syncthreads();
    float ex = __expf(e - mx);
    sv[tid] = ex;
    ss[tid] = ex;
    __syncthreads();
    for (int off = 64; off > 0; off >>= 1) {
        if (tid < off) ss[tid] += ss[tid + off];
        __syncthreads();
    }
    float inv = 1.f / ss[0];

    // ctx[d] = sum_s w[s] * enc[b,s,d]; also write bf16 into xg[:, 256:768]
    const float* eb = enc + (size_t)b * S2 * ENC;
    for (int d = tid; d < ENC; d += 128) {
        float a2 = 0.f;
        for (int s = 0; s < S2; ++s) a2 += sv[s] * eb[(size_t)s * ENC + d];
        a2 *= inv;
        ctx[b * ENC + d] = a2;
        xg[(size_t)b * (EMB + ENC) + EMB + d] = (bf16_t)a2;
    }
    for (int c = tid; c < EMB; c += 128)
        xg[(size_t)b * (EMB + ENC) + c] = emb_all[((size_t)b * TDEC + t) * EMB + c];
}

// ---------------------------------------------------------------------------
// Decoder GRU pointwise + [h|ctx] concat assembly.
// ---------------------------------------------------------------------------
__global__ void k_dec_pw(float* __restrict__ H, bf16_t* __restrict__ Hbf,
                         bf16_t* __restrict__ concat, const float* __restrict__ ctx,
                         const float* __restrict__ GI, const float* __restrict__ GH)
{
    int idx = blockIdx.x * blockDim.x + threadIdx.x;   // B * 768
    if (idx >= BSZ * (DH + ENC)) return;
    int b = idx / (DH + ENC);
    int j = idx - b * (DH + ENC);
    if (j < DH) {
        const float* gi = GI + (size_t)b * 3 * DH;
        const float* gh = GH + (size_t)b * 3 * DH;
        float ir = gi[j], iz = gi[DH + j], in_ = gi[2 * DH + j];
        float hr = gh[j], hz = gh[DH + j], hn  = gh[2 * DH + j];
        float rg = 1.f / (1.f + __expf(-(ir + hr)));
        float zg = 1.f / (1.f + __expf(-(iz + hz)));
        float ng = tanhf(in_ + rg * hn);
        float hp = H[b * DH + j];
        float hnew = (1.f - zg) * ng + zg * hp;
        H[b * DH + j] = hnew;
        Hbf[b * DH + j] = (bf16_t)hnew;
        concat[(size_t)b * (DH + ENC) + j] = (bf16_t)hnew;
    } else {
        concat[(size_t)b * (DH + ENC) + j] = (bf16_t)ctx[b * ENC + (j - DH)];
    }
}

// ---------------------------------------------------------------------------
// Host side
// ---------------------------------------------------------------------------
static inline void gemm(hipStream_t s, const bf16_t* A, int lda,
                        const bf16_t* W, int ldw, const float* bias,
                        float* Cf, int ldcf, bf16_t* Cb, int ldcb,
                        int M, int N, int Kd, int flags = 0,
                        const int* nl = nullptr, int Sdim = 1)
{
    int tiles = ((M + 31) / 32) * ((N + 31) / 32);
    int blocks = (tiles + 7) / 8;
    k_gemm<<<blocks, 256, 0, s>>>(A, lda, W, ldw, bias, Cf, ldcf, Cb, ldcb,
                                  M, N, Kd, flags, nl, Sdim);
}

static inline void cvt(hipStream_t s, bf16_t* dst, const float* src, int n)
{
    k_cvt<<<(n + 255) / 256, 256, 0, s>>>(dst, src, n);
}

extern "C" void kernel_launch(void* const* d_in, const int* in_sizes, int n_in,
                              void* d_out, int out_size, void* d_ws, size_t ws_size,
                              hipStream_t stream)
{
    // ---- inputs ----
    const float* src      = (const float*)d_in[0];
    const int*   lengths  = (const int*)d_in[1];
    const int*   tgt      = (const int*)d_in[2];
    const float* conv_w1  = (const float*)d_in[3];
    const float* conv_b1  = (const float*)d_in[4];
    const float* bn1_g    = (const float*)d_in[5];
    const float* bn1_b    = (const float*)d_in[6];
    const float* bn1_m    = (const float*)d_in[7];
    const float* bn1_v    = (const float*)d_in[8];
    const float* conv_w2  = (const float*)d_in[9];
    const float* conv_b2  = (const float*)d_in[10];
    const float* bn2_g    = (const float*)d_in[11];
    const float* bn2_b    = (const float*)d_in[12];
    const float* bn2_m    = (const float*)d_in[13];
    const float* bn2_v    = (const float*)d_in[14];
    const float* g_wih[4] = { (const float*)d_in[15], (const float*)d_in[19],
                              (const float*)d_in[23], (const float*)d_in[27] };
    const float* g_whh[4] = { (const float*)d_in[16], (const float*)d_in[20],
                              (const float*)d_in[24], (const float*)d_in[28] };
    const float* g_bih[4] = { (const float*)d_in[17], (const float*)d_in[21],
                              (const float*)d_in[25], (const float*)d_in[29] };
    const float* g_bhh[4] = { (const float*)d_in[18], (const float*)d_in[22],
                              (const float*)d_in[26], (const float*)d_in[30] };
    const float* proj_w   = (const float*)d_in[31];
    const float* proj_b   = (const float*)d_in[32];
    const float* ctc_w    = (const float*)d_in[33];
    const float* ctc_b    = (const float*)d_in[34];
    const float* dec_emb  = (const float*)d_in[35];
    const float* attn_enc_w = (const float*)d_in[36];
    const float* attn_dec_w = (const float*)d_in[37];
    const float* attn_v   = (const float*)d_in[38];
    const float* dec_wih  = (const float*)d_in[39];
    const float* dec_whh  = (const float*)d_in[40];
    const float* dec_bih  = (const float*)d_in[41];
    const float* dec_bhh  = (const float*)d_in[42];
    const float* out_w    = (const float*)d_in[43];
    const float* out_b    = (const float*)d_in[44];

    float* out_ctc = (float*)d_out;                         // [64,128,100]
    float* out_dec = (float*)d_out + BSZ * S2 * GV;         // [64,50,200]

    // ---- workspace bump allocator ----
    char* ws = (char*)d_ws;
    size_t off = 0;
    auto alloc = [&](size_t bytes) -> void* {
        void* p = ws + off;
        off = (off + bytes + 255) & ~(size_t)255;
        return p;
    };
    auto allocb = [&](size_t n) { return (bf16_t*)alloc(n * sizeof(bf16_t)); };
    auto allocf = [&](size_t n) { return (float*)alloc(n * sizeof(float)); };

    bf16_t* W1p = allocb((size_t)CC * K1PAD);   float* b1p = allocf(CC);
    bf16_t* W2p = allocb((size_t)CC * K2);      float* b2p = allocf(CC);
    bf16_t* wih_bf[4], *whh_bf[4];
    int wih_in[4] = { CC, CC, 2 * HH, 2 * HH };
    for (int d = 0; d < 4; ++d) {
        wih_bf[d] = allocb((size_t)3 * HH * wih_in[d]);
        whh_bf[d] = allocb((size_t)3 * HH * HH);
    }
    bf16_t* proj_bf = allocb((size_t)ENC * 2 * HH);
    bf16_t* ctc_bf  = allocb((size_t)GV * ENC);
    bf16_t* aenc_bf = allocb((size_t)ATT * ENC);
    bf16_t* adec_bf = allocb((size_t)ATT * DH);
    bf16_t* dwih_bf = allocb((size_t)3 * DH * (EMB + ENC));
    bf16_t* dwhh_bf = allocb((size_t)3 * DH * DH);
    bf16_t* outw_bf = allocb((size_t)TV * (DH + ENC));
    bf16_t* emb_bf  = allocb((size_t)TV * EMB);

    bf16_t* A1  = allocb((size_t)BSZ * S1 * K1PAD);
    bf16_t* X1  = allocb((size_t)BSZ * S1 * CC);
    bf16_t* A2  = allocb((size_t)BSZ * S2 * K2);
    bf16_t* X2  = allocb((size_t)BSZ * S2 * CC);
    float*  GIf = allocf((size_t)BSZ * S2 * 3 * HH);   // reused by both layers
    float*  GIb = allocf((size_t)BSZ * S2 * 3 * HH);
    float*  GHf = allocf((size_t)BSZ * 3 * HH);
    float*  GHb = allocf((size_t)BSZ * 3 * HH);
    float*  Hf  = allocf((size_t)BSZ * HH);
    float*  Hb  = allocf((size_t)BSZ * HH);
    bf16_t* Hfb = allocb((size_t)BSZ * HH);
    bf16_t* Hbb = allocb((size_t)BSZ * HH);
    bf16_t* h1     = allocb((size_t)BSZ * S2 * 2 * HH);
    bf16_t* encpre = allocb((size_t)BSZ * S2 * 2 * HH);
    float*  encf   = allocf((size_t)BSZ * S2 * ENC);
    bf16_t* encbf  = allocb((size_t)BSZ * S2 * ENC);
    float*  ENCP   = allocf((size_t)BSZ * S2 * ATT);
    int*    nl     = (int*)alloc(BSZ * sizeof(int));
    bf16_t* emb_all = allocb((size_t)BSZ * TDEC * EMB);
    float*  DP   = allocf((size_t)BSZ * ATT);
    bf16_t* xg   = allocb((size_t)BSZ * (EMB + ENC));
    float*  ctx  = allocf((size_t)BSZ * ENC);
    float*  GID  = allocf((size_t)BSZ * 3 * DH);
    float*  GHD  = allocf((size_t)BSZ * 3 * DH);
    float*  Hd   = allocf((size_t)BSZ * DH);
    bf16_t* Hdb  = allocb((size_t)BSZ * DH);
    bf16_t* cat  = allocb((size_t)BSZ * (DH + ENC));
    (void)ws_size; (void)n_in; (void)in_sizes; (void)out_size;

    // ---- prep: fold BN + convert weights to bf16 ----
    k_fold_conv<<<CC, 256, 0, stream>>>(W1p, b1p, conv_w1, conv_b1, bn1_g, bn1_b,
                                        bn1_m, bn1_v, DIN, K1PAD);
    k_fold_conv<<<CC, 256, 0, stream>>>(W2p, b2p, conv_w2, conv_b2, bn2_g, bn2_b,
                                        bn2_m, bn2_v, CC, K2);
    for (int d = 0; d < 4; ++d) {
        cvt(stream, wih_bf[d], g_wih[d], 3 * HH * wih_in[d]);
        cvt(stream, whh_bf[d], g_whh[d], 3 * HH * HH);
    }
    cvt(stream, proj_bf, proj_w, ENC * 2 * HH);
    cvt(stream, ctc_bf,  ctc_w,  GV * ENC);
    cvt(stream, aenc_bf, attn_enc_w, ATT * ENC);
    cvt(stream, adec_bf, attn_dec_w, ATT * DH);
    cvt(stream, dwih_bf, dec_wih, 3 * DH * (EMB + ENC));
    cvt(stream, dwhh_bf, dec_whh, 3 * DH * DH);
    cvt(stream, outw_bf, out_w, TV * (DH + ENC));
    cvt(stream, emb_bf,  dec_emb, TV * EMB);
    k_nl<<<1, 64, 0, stream>>>(nl, lengths);
    k_emb<<<(BSZ * TDEC * EMB + 255) / 256, 256, 0, stream>>>(emb_all, tgt, emb_bf);

    // ---- convs as im2col + WMMA GEMM (BN folded, ReLU in epilogue) ----
    k_im2col1<<<BSZ * S1, 256, 0, stream>>>(A1, src);
    gemm(stream, A1, K1PAD, W1p, K1PAD, b1p, nullptr, 0, X1, CC,
         BSZ * S1, CC, K1PAD, GF_RELU);
    k_im2col2<<<BSZ * S2, 256, 0, stream>>>(A2, X1);
    gemm(stream, A2, K2, W2p, K2, b2p, nullptr, 0, X2, CC,
         BSZ * S2, CC, K2, GF_RELU);

    // ---- BiGRU layer 1: hoisted input GEMMs, then 128 recurrent steps ----
    gemm(stream, X2, CC, wih_bf[0], CC, g_bih[0], GIf, 3 * HH, nullptr, 0,
         BSZ * S2, 3 * HH, CC);
    gemm(stream, X2, CC, wih_bf[1], CC, g_bih[1], GIb, 3 * HH, nullptr, 0,
         BSZ * S2, 3 * HH, CC);
    k_zero_f32<<<(BSZ * HH + 255) / 256, 256, 0, stream>>>(Hf, BSZ * HH);
    k_zero_f32<<<(BSZ * HH + 255) / 256, 256, 0, stream>>>(Hb, BSZ * HH);
    k_zero_bf<<<(BSZ * HH + 255) / 256, 256, 0, stream>>>(Hfb, BSZ * HH);
    k_zero_bf<<<(BSZ * HH + 255) / 256, 256, 0, stream>>>(Hbb, BSZ * HH);
    for (int t = 0; t < S2; ++t) {
        gemm(stream, Hfb, HH, whh_bf[0], HH, g_bhh[0], GHf, 3 * HH, nullptr, 0,
             BSZ, 3 * HH, HH);
        gemm(stream, Hbb, HH, whh_bf[1], HH, g_bhh[1], GHb, 3 * HH, nullptr, 0,
             BSZ, 3 * HH, HH);
        k_gru_pw<<<(2 * BSZ * HH + 255) / 256, 256, 0, stream>>>(
            Hf, Hb, Hfb, Hbb, h1, GIf, GIb, GHf, GHb, t);
    }

    // ---- BiGRU layer 2 ----
    gemm(stream, h1, 2 * HH, wih_bf[2], 2 * HH, g_bih[2], GIf, 3 * HH, nullptr, 0,
         BSZ * S2, 3 * HH, 2 * HH);
    gemm(stream, h1, 2 * HH, wih_bf[3], 2 * HH, g_bih[3], GIb, 3 * HH, nullptr, 0,
         BSZ * S2, 3 * HH, 2 * HH);
    k_zero_f32<<<(BSZ * HH + 255) / 256, 256, 0, stream>>>(Hf, BSZ * HH);
    k_zero_f32<<<(BSZ * HH + 255) / 256, 256, 0, stream>>>(Hb, BSZ * HH);
    k_zero_bf<<<(BSZ * HH + 255) / 256, 256, 0, stream>>>(Hfb, BSZ * HH);
    k_zero_bf<<<(BSZ * HH + 255) / 256, 256, 0, stream>>>(Hbb, BSZ * HH);
    for (int t = 0; t < S2; ++t) {
        gemm(stream, Hfb, HH, whh_bf[2], HH, g_bhh[2], GHf, 3 * HH, nullptr, 0,
             BSZ, 3 * HH, HH);
        gemm(stream, Hbb, HH, whh_bf[3], HH, g_bhh[3], GHb, 3 * HH, nullptr, 0,
             BSZ, 3 * HH, HH);
        k_gru_pw<<<(2 * BSZ * HH + 255) / 256, 256, 0, stream>>>(
            Hf, Hb, Hfb, Hbb, encpre, GIf, GIb, GHf, GHb, t);
    }

    // ---- projection (+length mask), CTC head, attention encoder projection ----
    gemm(stream, encpre, 2 * HH, proj_bf, 2 * HH, proj_b, encf, ENC, encbf, ENC,
         BSZ * S2, ENC, 2 * HH, GF_MASK, nl, S2);
    gemm(stream, encbf, ENC, ctc_bf, ENC, ctc_b, out_ctc, GV, nullptr, 0,
         BSZ * S2, GV, ENC);
    gemm(stream, encbf, ENC, aenc_bf, ENC, nullptr, ENCP, ATT, nullptr, 0,
         BSZ * S2, ATT, ENC);

    // ---- attention GRU decoder, 50 steps ----
    k_zero_f32<<<(BSZ * DH + 255) / 256, 256, 0, stream>>>(Hd, BSZ * DH);
    k_zero_bf<<<(BSZ * DH + 255) / 256, 256, 0, stream>>>(Hdb, BSZ * DH);
    for (int t = 0; t < TDEC; ++t) {
        gemm(stream, Hdb, DH, adec_bf, DH, nullptr, DP, ATT, nullptr, 0,
             BSZ, ATT, DH);
        k_attn<<<BSZ, 128, 0, stream>>>(ENCP, DP, encf, attn_v, nl, emb_all,
                                        xg, ctx, t);
        gemm(stream, xg, EMB + ENC, dwih_bf, EMB + ENC, dec_bih, GID, 3 * DH,
             nullptr, 0, BSZ, 3 * DH, EMB + ENC);
        gemm(stream, Hdb, DH, dwhh_bf, DH, dec_bhh, GHD, 3 * DH,
             nullptr, 0, BSZ, 3 * DH, DH);
        k_dec_pw<<<(BSZ * (DH + ENC) + 255) / 256, 256, 0, stream>>>(
            Hd, Hdb, cat, ctx, GID, GHD);
        gemm(stream, cat, DH + ENC, outw_bf, DH + ENC, out_b,
             out_dec + t * TV, TDEC * TV, nullptr, 0, BSZ, TV, DH + ENC);
    }
}
